// EncoderForSequenceClassification_9234179687197
// MI455X (gfx1250) — matrix operand order repaired and checked by reading, not compiled, not run on hardware
//
#include <hip/hip_runtime.h>
#include <hip/hip_bf16.h>

#define Bsz   16
#define Ssz   512
#define Dsz   768
#define Hsz   12
#define Fsz   3072
#define LYRn  12
#define DHsz  64
#define BStok (Bsz*Ssz)   // 8192 token rows

typedef _Float16 half16v __attribute__((ext_vector_type(16)));
typedef _Float16 half8v  __attribute__((ext_vector_type(8)));
typedef _Float16 half4v  __attribute__((ext_vector_type(4)));
typedef float    float8v __attribute__((ext_vector_type(8)));

#if defined(__HIP_DEVICE_COMPILE__) &&                                      \
    __has_builtin(__builtin_amdgcn_global_load_async_to_lds_b128) &&        \
    __has_builtin(__builtin_amdgcn_s_wait_asynccnt)
#define USE_ASYNC_LDS 1
#else
#define USE_ASYNC_LDS 0
#endif

#if USE_ASYNC_LDS
typedef int int4vs __attribute__((vector_size(4 * sizeof(int))));
typedef __attribute__((address_space(1))) int4vs* gptr4_t;
typedef __attribute__((address_space(3))) int4vs* lptr4_t;
__device__ __forceinline__ void async_cp16(void* lds, const void* g) {
  __builtin_amdgcn_global_load_async_to_lds_b128(
      (gptr4_t)(void*)g, (lptr4_t)lds, 0, 0);
}
#endif

// ---- WMMA fragment loaders (per CDNA5 ISA VGPR layouts, wave32) ----
// A matrix 16x32 f16: lanes 0-15 hold K=[0..7],[16..23]; lanes 16-31 K=[8..15],[24..31]
__device__ __forceinline__ half16v ld_frag_a(const _Float16* rowbase, int hi) {
  half8v lo = *(const half8v*)(rowbase + 8 * hi);
  half8v up = *(const half8v*)(rowbase + 16 + 8 * hi);
  return __builtin_shufflevector(lo, up, 0,1,2,3,4,5,6,7,8,9,10,11,12,13,14,15);
}
// B matrix 32x16 f16: lanes 0-15 hold K=[0..7],[16..23]; lanes 16-31 K=[8..15],[24..31]
__device__ __forceinline__ half16v ld_frag_b(const _Float16* colbase, int hi) {
  half8v lo = *(const half8v*)(colbase + 8 * hi);
  half8v up = *(const half8v*)(colbase + 16 + 8 * hi);
  return __builtin_shufflevector(lo, up, 0,1,2,3,4,5,6,7,8,9,10,11,12,13,14,15);
}
__device__ __forceinline__ float8v wmma_f16(half16v a, half16v b, float8v c) {
  return __builtin_amdgcn_wmma_f32_16x16x32_f16(false, a, false, b, (short)0, c,
                                                false, false);
}

// =====================================================================
// Embedding gather: tmp[row][d] = token_emb[ids[row]][d] + pos_emb[s][d]
// =====================================================================
__global__ __launch_bounds__(256) void emb_kernel(
    const int* __restrict__ ids, const float* __restrict__ tok,
    const float* __restrict__ pos, float* __restrict__ out) {
  int row = blockIdx.x;
  int s = row % Ssz;
  int id = ids[row];
  const float* tr = tok + (size_t)id * Dsz;
  const float* pr = pos + (size_t)s * Dsz;
  float* orow = out + (size_t)row * Dsz;
  for (int d = threadIdx.x; d < Dsz; d += 256) orow[d] = tr[d] + pr[d];
}

// =====================================================================
// LayerNorm over D=768, one block per row. Writes f32 and/or f16.
// =====================================================================
__global__ __launch_bounds__(256) void ln_kernel(
    const float* __restrict__ x, const float* __restrict__ scale,
    const float* __restrict__ bias, float* __restrict__ out_f32,
    _Float16* __restrict__ out_f16) {
  int row = blockIdx.x;
  int tid = threadIdx.x, lane = tid & 31, wave = tid >> 5;
  const float* xr = x + (size_t)row * Dsz;
  float vals[3];
  float s = 0.f, s2 = 0.f;
#pragma unroll
  for (int i = 0; i < 3; ++i) {
    float v = xr[tid + i * 256];
    vals[i] = v; s += v; s2 += v * v;
  }
#pragma unroll
  for (int off = 16; off > 0; off >>= 1) {
    s += __shfl_down(s, off);
    s2 += __shfl_down(s2, off);
  }
  __shared__ float a1[8], a2[8], stats[2];
  if (lane == 0) { a1[wave] = s; a2[wave] = s2; }
  __syncthreads();
  if (tid == 0) {
    float t1 = 0.f, t2 = 0.f;
#pragma unroll
    for (int w = 0; w < 8; ++w) { t1 += a1[w]; t2 += a2[w]; }
    float mu = t1 * (1.0f / Dsz);
    float var = t2 * (1.0f / Dsz) - mu * mu;
    stats[0] = mu;
    stats[1] = rsqrtf(var + 1e-5f);
  }
  __syncthreads();
  float mu = stats[0], rs = stats[1];
#pragma unroll
  for (int i = 0; i < 3; ++i) {
    int d = tid + i * 256;
    float y = (vals[i] - mu) * rs * scale[d] + bias[d];
    if (out_f32) out_f32[(size_t)row * Dsz + d] = y;
    if (out_f16) out_f16[(size_t)row * Dsz + d] = (_Float16)y;
  }
}

// =====================================================================
// Weight fp32 -> f16 with transpose: Wt[n][k] = W[k][n]   (K x N source)
// =====================================================================
__global__ __launch_bounds__(256) void convert_wt_plain(
    const float* __restrict__ W, _Float16* __restrict__ Wt, int K, int N) {
  int n = blockIdx.x;
  for (int k = threadIdx.x; k < K; k += 256)
    Wt[(size_t)n * K + k] = (_Float16)W[(size_t)k * N + n];
}
// wq layout (H, D, DH) -> Wt[n=(h*64+e)][k=d]
__global__ __launch_bounds__(256) void convert_wt_qkv(
    const float* __restrict__ W, _Float16* __restrict__ Wt) {
  int n = blockIdx.x;           // 0..767
  int h = n >> 6, e = n & 63;
  for (int k = threadIdx.x; k < Dsz; k += 256)
    Wt[(size_t)n * Dsz + k] = (_Float16)W[((size_t)h * Dsz + k) * DHsz + e];
}

// =====================================================================
// Tiled WMMA GEMM: C[M,N] = A[M,K](f16) * Bt[N,K](f16)  + epilogue
// block tile 128x128, 8 waves (4 M x 2 N), each wave 32x64 (2x4 WMMA
// tiles), K step 32.  Staging via GLOBAL_LOAD_ASYNC_TO_LDS when available.
// EPI 0: store f16.  EPI 1: xres = xres + acc + bias (f32, in-place).
// EPI 2: store f16 gelu(acc + bias).
// =====================================================================
template <int EPI>
__global__ __launch_bounds__(256) void gemm_kernel(
    const _Float16* __restrict__ A, const _Float16* __restrict__ Bt,
    const float* __restrict__ bias, float* __restrict__ xres,
    _Float16* __restrict__ outh, int M, int N, int K) {
  __shared__ __align__(16) _Float16 As[128][40];
  __shared__ __align__(16) _Float16 Bs[128][40];

  const int tid = threadIdx.x;
  const int lane = tid & 31, wave = tid >> 5;
  const int l15 = lane & 15, hi = lane >> 4;
  const int wm = (wave >> 1) * 32;   // 0,32,64,96
  const int wn = (wave & 1) * 64;    // 0,64
  const int m0 = blockIdx.y * 128, n0 = blockIdx.x * 128;

  const int sr = tid >> 1, sc0 = (tid & 1) * 16;  // staging: 2 x half8 each

  float8v acc[2][4];
#pragma unroll
  for (int i = 0; i < 2; ++i)
#pragma unroll
    for (int j = 0; j < 4; ++j)
      acc[i][j] = (float8v){0.f,0.f,0.f,0.f,0.f,0.f,0.f,0.f};

  const int nk = K >> 5;
  for (int kc = 0; kc < nk; ++kc) {
    __syncthreads();
    const _Float16* ap = A + (size_t)(m0 + sr) * K + kc * 32 + sc0;
    const _Float16* bp = Bt + (size_t)(n0 + sr) * K + kc * 32 + sc0;
#if USE_ASYNC_LDS
    async_cp16(&As[sr][sc0], ap);
    async_cp16(&As[sr][sc0 + 8], ap + 8);
    async_cp16(&Bs[sr][sc0], bp);
    async_cp16(&Bs[sr][sc0 + 8], bp + 8);
    __builtin_amdgcn_s_wait_asynccnt(0);
#else
    *(half8v*)&As[sr][sc0] = *(const half8v*)ap;
    *(half8v*)&As[sr][sc0 + 8] = *(const half8v*)(ap + 8);
    *(half8v*)&Bs[sr][sc0] = *(const half8v*)bp;
    *(half8v*)&Bs[sr][sc0 + 8] = *(const half8v*)(bp + 8);
#endif
    __syncthreads();

    if (kc + 1 < nk) {  // prefetch next K-chunk into cache
      __builtin_prefetch(ap + 32, 0, 0);
      __builtin_prefetch(bp + 32, 0, 0);
    }

    half16v a0 = ld_frag_a(&As[wm + l15][0], hi);
    half16v a1 = ld_frag_a(&As[wm + 16 + l15][0], hi);
#pragma unroll
    for (int j = 0; j < 4; ++j) {
      half16v bj = ld_frag_b(&Bs[wn + j * 16 + l15][0], hi);
      acc[0][j] = wmma_f16(a0, bj, acc[0][j]);
      acc[1][j] = wmma_f16(a1, bj, acc[1][j]);
    }
  }

#pragma unroll
  for (int i = 0; i < 2; ++i)
#pragma unroll
    for (int j = 0; j < 4; ++j) {
      int n = n0 + wn + j * 16 + l15;
      float bval = (EPI != 0 && bias) ? bias[n] : 0.f;
#pragma unroll
      for (int v = 0; v < 8; ++v) {
        int m = m0 + wm + i * 16 + hi * 8 + v;
        float val = acc[i][j][v];
        size_t idx = (size_t)m * N + n;
        if constexpr (EPI == 0) {
          outh[idx] = (_Float16)val;
        } else if constexpr (EPI == 1) {
          xres[idx] = xres[idx] + val + bval;
        } else {
          float t = val + bval;
          float g = 0.5f * t * (1.0f + erff(t * 0.70710678118654752f));
          outh[idx] = (_Float16)g;
        }
      }
    }
}

// =====================================================================
// Attention: one block per (b*H+h, 16-query-row slab). log_softmax!
// =====================================================================
__global__ __launch_bounds__(256) void attn_kernel(
    const _Float16* __restrict__ q, const _Float16* __restrict__ k,
    const _Float16* __restrict__ v, _Float16* __restrict__ ctx) {
  __shared__ __align__(16) _Float16 qs[16][72];
  __shared__ float sc[16][520];
  __shared__ __align__(16) _Float16 wf[16][520];
  __shared__ __align__(16) _Float16 vt[64][40];
  __shared__ float red[16][17];
  __shared__ float lse[16];

  const int bh = blockIdx.x;
  const int b = bh / Hsz, h = bh % Hsz;
  const int qb = blockIdx.y;  // 16 query rows per block
  const int tid = threadIdx.x;
  const int lane = tid & 31, wave = tid >> 5;
  const int l15 = lane & 15, hi = lane >> 4;

  // ---- stage Q slab (16 x 64) ----
  {
    int r = tid >> 4, c0 = (tid & 15) * 4;
    const _Float16* src =
        q + ((size_t)(b * Ssz + qb * 16 + r)) * Dsz + h * DHsz + c0;
    *(half4v*)&qs[r][c0] = *(const half4v*)src;
  }
  __syncthreads();

  half16v qa0 = ld_frag_a(&qs[l15][0], hi);
  half16v qa1 = ld_frag_a(&qs[l15][32], hi);

  // ---- pass 1: scores = (Q K^T) / 8, tiles of 16 keys, 4 per wave ----
  const _Float16* kbase = k + ((size_t)b * Ssz) * Dsz + h * DHsz;
#pragma unroll
  for (int c = 0; c < 4; ++c) {
    int ct = wave * 4 + c;
    const _Float16* krow = kbase + (size_t)(ct * 16 + l15) * Dsz;
    half16v kb0 = ld_frag_b(krow, hi);
    half16v kb1 = ld_frag_b(krow + 32, hi);
    float8v a = (float8v){0.f,0.f,0.f,0.f,0.f,0.f,0.f,0.f};
    a = wmma_f16(qa0, kb0, a);
    a = wmma_f16(qa1, kb1, a);
#pragma unroll
    for (int vv = 0; vv < 8; ++vv)
      sc[hi * 8 + vv][ct * 16 + l15] = a[vv] * 0.125f;
  }
  __syncthreads();

  // ---- pass 2: row-wise log-sum-exp over 512 ----
  {
    int r = tid >> 4, c = tid & 15;
    float m = -3.4e38f;
    for (int t = c; t < Ssz; t += 16) m = fmaxf(m, sc[r][t]);
    red[r][c] = m;
    __syncthreads();
    if (c == 0) {
      float mm = red[r][0];
#pragma unroll
      for (int j = 1; j < 16; ++j) mm = fmaxf(mm, red[r][j]);
      red[r][16] = mm;
    }
    __syncthreads();
    float rowmax = red[r][16];
    float s = 0.f;
    for (int t = c; t < Ssz; t += 16) s += expf(sc[r][t] - rowmax);
    __syncthreads();
    red[r][c] = s;
    __syncthreads();
    if (c == 0) {
      float ss = 0.f;
#pragma unroll
      for (int j = 0; j < 16; ++j) ss += red[r][j];
      lse[r] = rowmax + logf(ss);
    }
  }
  __syncthreads();

  // ---- pass 2b: w = log_softmax(scores) as f16 ----
  for (int i = tid; i < 16 * Ssz; i += 256) {
    int r = i >> 9, t = i & 511;
    wf[r][t] = (_Float16)(sc[r][t] - lse[r]);
  }
  __syncthreads();

  // ---- pass 3: ctx = w @ V  (waves 0-3, one 16x16 dh-tile each) ----
  float8v cacc = (float8v){0.f,0.f,0.f,0.f,0.f,0.f,0.f,0.f};
  const _Float16* vbase = v + ((size_t)b * Ssz) * Dsz + h * DHsz;
  for (int kc = 0; kc < Ssz / 32; ++kc) {
    __syncthreads();
    {  // stage V chunk transposed: vt[dh][t]
      int t = tid >> 3, d0 = (tid & 7) * 8;
      half8v vv = *(const half8v*)(vbase + (size_t)(kc * 32 + t) * Dsz + d0);
#pragma unroll
      for (int j = 0; j < 8; ++j) vt[d0 + j][t] = vv[j];
    }
    __syncthreads();
    if (wave < 4) {
      half16v wa = ld_frag_a(&wf[l15][kc * 32], hi);
      half16v vb = ld_frag_b(&vt[wave * 16 + l15][0], hi);
      cacc = wmma_f16(wa, vb, cacc);
    }
  }
  if (wave < 4) {
#pragma unroll
    for (int vv = 0; vv < 8; ++vv) {
      int s = qb * 16 + hi * 8 + vv;
      int col = h * DHsz + wave * 16 + l15;
      ctx[((size_t)(b * Ssz + s)) * Dsz + col] = (_Float16)cacc[vv];
    }
  }
}

// =====================================================================
// Classifier: out[b][n] = h[b,0,:] . clf_w[:,n] + clf_b[n]   (16x2)
// =====================================================================
__global__ __launch_bounds__(256) void clf_kernel(
    const _Float16* __restrict__ hfin, const float* __restrict__ w,
    const float* __restrict__ bias, float* __restrict__ out) {
  int tid = threadIdx.x;
  int p = tid >> 3;       // 32 (b,n) pairs
  int b = p >> 1, n = p & 1, l8 = tid & 7;
  const _Float16* hr = hfin + ((size_t)b * Ssz) * Dsz;
  float s = 0.f;
  for (int d = l8; d < Dsz; d += 8) s += (float)hr[d] * w[d * 2 + n];
  s += __shfl_down(s, 4, 8);
  s += __shfl_down(s, 2, 8);
  s += __shfl_down(s, 1, 8);
  if (l8 == 0) out[b * 2 + n] = s + bias[n];
}

// =====================================================================
extern "C" void kernel_launch(void* const* d_in, const int* in_sizes, int n_in,
                              void* d_out, int out_size, void* d_ws,
                              size_t ws_size, hipStream_t stream) {
  const int* input_ids   = (const int*)d_in[0];
  const float* token_emb = (const float*)d_in[1];
  const float* pos_emb   = (const float*)d_in[2];
  const float* emb_ln_s  = (const float*)d_in[3];
  const float* emb_ln_b  = (const float*)d_in[4];
  const float* wq        = (const float*)d_in[5];
  const float* wk        = (const float*)d_in[6];
  const float* wv        = (const float*)d_in[7];
  const float* wo        = (const float*)d_in[8];
  const float* wo_b      = (const float*)d_in[9];
  const float* ln1_s     = (const float*)d_in[10];
  const float* ln1_b     = (const float*)d_in[11];
  const float* ln2_s     = (const float*)d_in[12];
  const float* ln2_b     = (const float*)d_in[13];
  const float* ff1_w     = (const float*)d_in[14];
  const float* ff1_b     = (const float*)d_in[15];
  const float* ff2_w     = (const float*)d_in[16];
  const float* ff2_b     = (const float*)d_in[17];
  const float* fin_ln_s  = (const float*)d_in[18];
  const float* fin_ln_b  = (const float*)d_in[19];
  const float* clf_w     = (const float*)d_in[20];
  const float* clf_b     = (const float*)d_in[21];
  float* out = (float*)d_out;

  char* ws = (char*)d_ws;
  size_t off = 0;
  auto alloc = [&](size_t bytes) -> void* {
    void* p = ws + off;
    off += (bytes + 255) & ~(size_t)255;
    return p;
  };
  float*     tmp  = (float*)alloc((size_t)BStok * Dsz * 4);
  float*     x    = (float*)alloc((size_t)BStok * Dsz * 4);
  _Float16*  hb   = (_Float16*)alloc((size_t)BStok * Dsz * 2);
  _Float16*  qb   = (_Float16*)alloc((size_t)BStok * Dsz * 2);
  _Float16*  kb   = (_Float16*)alloc((size_t)BStok * Dsz * 2);
  _Float16*  vb   = (_Float16*)alloc((size_t)BStok * Dsz * 2);
  _Float16*  ctxb = (_Float16*)alloc((size_t)BStok * Dsz * 2);
  _Float16*  ffb  = (_Float16*)alloc((size_t)BStok * Fsz * 2);
  _Float16*  wq16 = (_Float16*)alloc((size_t)Dsz * Dsz * 2);
  _Float16*  wk16 = (_Float16*)alloc((size_t)Dsz * Dsz * 2);
  _Float16*  wv16 = (_Float16*)alloc((size_t)Dsz * Dsz * 2);
  _Float16*  wo16 = (_Float16*)alloc((size_t)Dsz * Dsz * 2);
  _Float16*  f116 = (_Float16*)alloc((size_t)Dsz * Fsz * 2);
  _Float16*  f216 = (_Float16*)alloc((size_t)Fsz * Dsz * 2);

  const dim3 blk(256);
  const dim3 gD(Dsz / 128, BStok / 128);   // N=768 GEMMs
  const dim3 gF(Fsz / 128, BStok / 128);   // N=3072 GEMM

  emb_kernel<<<BStok, blk, 0, stream>>>(input_ids, token_emb, pos_emb, tmp);
  ln_kernel<<<BStok, blk, 0, stream>>>(tmp, emb_ln_s, emb_ln_b, x, nullptr);

  for (int l = 0; l < LYRn; ++l) {
    ln_kernel<<<BStok, blk, 0, stream>>>(x, ln1_s + l * Dsz, ln1_b + l * Dsz,
                                         nullptr, hb);
    convert_wt_qkv<<<Dsz, blk, 0, stream>>>(wq + (size_t)l * Hsz * Dsz * DHsz, wq16);
    convert_wt_qkv<<<Dsz, blk, 0, stream>>>(wk + (size_t)l * Hsz * Dsz * DHsz, wk16);
    convert_wt_qkv<<<Dsz, blk, 0, stream>>>(wv + (size_t)l * Hsz * Dsz * DHsz, wv16);
    gemm_kernel<0><<<gD, blk, 0, stream>>>(hb, wq16, nullptr, nullptr, qb,
                                           BStok, Dsz, Dsz);
    gemm_kernel<0><<<gD, blk, 0, stream>>>(hb, wk16, nullptr, nullptr, kb,
                                           BStok, Dsz, Dsz);
    gemm_kernel<0><<<gD, blk, 0, stream>>>(hb, wv16, nullptr, nullptr, vb,
                                           BStok, Dsz, Dsz);
    attn_kernel<<<dim3(Bsz * Hsz, Ssz / 16), blk, 0, stream>>>(qb, kb, vb, ctxb);
    convert_wt_plain<<<Dsz, blk, 0, stream>>>(wo + (size_t)l * Dsz * Dsz, wo16,
                                              Dsz, Dsz);
    gemm_kernel<1><<<gD, blk, 0, stream>>>(ctxb, wo16, wo_b + l * Dsz, x,
                                           nullptr, BStok, Dsz, Dsz);
    ln_kernel<<<BStok, blk, 0, stream>>>(x, ln2_s + l * Dsz, ln2_b + l * Dsz,
                                         nullptr, hb);
    convert_wt_plain<<<Fsz, blk, 0, stream>>>(ff1_w + (size_t)l * Dsz * Fsz,
                                              f116, Dsz, Fsz);
    gemm_kernel<2><<<gF, blk, 0, stream>>>(hb, f116, ff1_b + l * Fsz, nullptr,
                                           ffb, BStok, Fsz, Dsz);
    convert_wt_plain<<<Dsz, blk, 0, stream>>>(ff2_w + (size_t)l * Fsz * Dsz,
                                              f216, Fsz, Dsz);
    gemm_kernel<1><<<gD, blk, 0, stream>>>(ffb, f216, ff2_b + l * Dsz, x,
                                           nullptr, BStok, Dsz, Fsz);
  }

  ln_kernel<<<BStok, blk, 0, stream>>>(x, fin_ln_s, fin_ln_b, nullptr, hb);
  clf_kernel<<<1, blk, 0, stream>>>(hb, clf_w, clf_b, out);
}
